// AttentionDecoder_31250182046330
// MI455X (gfx1250) — matrix operand order repaired and checked by reading, not compiled, not run on hardware
//
#include <hip/hip_runtime.h>
#include <hip/hip_bf16.h>

// ---------------------------------------------------------------------------
// AttentionDecoder forward for MI455X (gfx1250): wave32, WMMA f16->f32,
// Tensor Data Mover (tensor_load_to_lds) for GEMM operand staging.
// B=64, S=128, SRC=256, H=512, V=8000, NH=8, HD=64.
// ---------------------------------------------------------------------------

#define BB   64
#define SS   128
#define SRCL 256
#define HH   512
#define VV   8000
#define NHH  8
#define HDD  64

typedef _Float16 v4h_t  __attribute__((ext_vector_type(4)));
typedef _Float16 v8h_t  __attribute__((ext_vector_type(8)));
typedef _Float16 v16h_t __attribute__((ext_vector_type(16)));
typedef float    v8f_t  __attribute__((ext_vector_type(8)));
typedef unsigned int u32x4 __attribute__((ext_vector_type(4)));
typedef int          i32x8 __attribute__((ext_vector_type(8)));
typedef int          i32x4 __attribute__((ext_vector_type(4)));

// ---------------------------------------------------------------------------
// TDM 2D tile load: global f16 tile (tile_rows x tile_k elements, row stride
// stride_elems) -> LDS at lds_addr with 16B padding after every 128B row
// (matching the [rows][72] _Float16 LDS layout).
// D# packing per CDNA5 ISA.
// ---------------------------------------------------------------------------
__device__ __forceinline__ void tdm_load_2d(unsigned lds_addr, const _Float16* gptr,
                                            unsigned tile_k, unsigned tile_rows,
                                            unsigned tdim0, unsigned tdim1,
                                            unsigned long stride_elems)
{
    unsigned long ga = (unsigned long)gptr;
    u32x4 g0;
    g0[0] = 1u;                                   // count=1, user descriptor
    g0[1] = lds_addr;                             // lds_addr (bytes)
    g0[2] = (unsigned)(ga & 0xffffffffu);         // global_addr[31:0]
    g0[3] = (unsigned)((ga >> 32) & 0x01ffffffu)  // global_addr[56:32]
          | (2u << 30);                           // type = 2 ("image")
    i32x8 g1;
    unsigned d0 = (1u << 16)                      // data_size = 1 -> 2 bytes
                | (1u << 20)                      // pad_enable
                | (4u << 22)                      // pad_interval: 32 DWORDs (128B)
                | (3u << 25);                     // pad_amount: 4 DWORDs (16B)
    g1[0] = (int)d0;
    g1[1] = (int)((tdim0 & 0xffffu) << 16);                       // tensor_dim0 lo16
    g1[2] = (int)((tdim0 >> 16) | ((tdim1 & 0xffffu) << 16));     // dim0 hi | dim1 lo
    g1[3] = (int)((tdim1 >> 16) | (tile_k << 16));                // dim1 hi | tile_dim0
    g1[4] = (int)(tile_rows & 0xffffu);                           // tile_dim1 (dim2=0)
    g1[5] = (int)(unsigned)(stride_elems & 0xffffffffu);          // dim0_stride lo32
    g1[6] = (int)(unsigned)((stride_elems >> 32) & 0xffffu);      // dim0_stride hi16
    g1[7] = 0;
    i32x4 z4 = {0, 0, 0, 0};
    i32x8 z8 = {0, 0, 0, 0, 0, 0, 0, 0};
    __builtin_amdgcn_tensor_load_to_lds(g0, g1, z4, z4, z8, 0);
}

// ---------------------------------------------------------------------------
// Strided-batched GEMM on f16 operands: C[z](MxN) = alpha*A*B + bias
//   A: row-major K-contiguous f16:  A[zA + m*lar + k]
//   B: lbc==1 ("NT"): B[zB + n*lbr + k] (TDM-staged)
//      else   ("NN"): B[zB + k*lbc + n] (manual f16 transpose stage)
//   C: f32 or f16 (c_half), row-major, stride ldc.
// Block 256 thr / 8 waves; tile 128x64; K chunk 64; 8 WMMAs per wave per chunk.
// Pipeline: wave0/wave1 issue TDM loads for the next LDS buffer, all waves
// compute on the current one; s_wait_tensorcnt + one barrier per chunk.
// ---------------------------------------------------------------------------
__global__ __launch_bounds__(256)
void k_gemm_wmma(const _Float16* __restrict__ A, const _Float16* __restrict__ Bm,
                 const float* __restrict__ bias, void* __restrict__ Cp, int c_half,
                 int M, int Nn, int K, long lar, long lbr, long lbc, long ldc,
                 int binner, long sAo, long sAi, long sBo, long sBi,
                 long sCo, long sCi, float alpha)
{
    __shared__ __align__(16) _Float16 As[2][128][72];
    __shared__ __align__(16) _Float16 Bs[2][64][72];

    const int t = threadIdx.x;
    const int z = blockIdx.z;
    const long zo = z / binner, zi = z % binner;
    const _Float16* __restrict__ Ab = A  + zo * sAo + zi * sAi;
    const _Float16* __restrict__ Bb = Bm + zo * sBo + zi * sBi;
    const long zC = zo * sCo + zi * sCi;
    const long m0 = (long)blockIdx.y * 128;
    const long n0 = (long)blockIdx.x * 64;

    const int wave = t >> 5, lane = t & 31;
    const int wm = wave >> 2, wn = wave & 3;     // 2 x 4 wave grid
    const int c16 = lane & 15, kh = lane >> 4;
    const bool bnt = (lbc == 1);

    // Unprovable dead store: keeps the TDM-written LDS from looking undef.
    if (M == -123456789) { As[0][0][0] = (_Float16)0; Bs[0][0][0] = (_Float16)0; }

    const unsigned ldsA0 = (unsigned)(unsigned long)&As[0][0][0];
    const unsigned ldsA1 = (unsigned)(unsigned long)&As[1][0][0];
    const unsigned ldsB0 = (unsigned)(unsigned long)&Bs[0][0][0];
    const unsigned ldsB1 = (unsigned)(unsigned long)&Bs[1][0][0];

    auto issueA = [&](int buf, int kc) {
        tdm_load_2d(buf ? ldsA1 : ldsA0, Ab + m0 * lar + kc, 64u, 128u,
                    (unsigned)(K - kc), (unsigned)(M - m0), (unsigned long)lar);
    };
    auto issueB = [&](int buf, int kc) {
        tdm_load_2d(buf ? ldsB1 : ldsB0, Bb + n0 * lbr + kc, 64u, 64u,
                    (unsigned)(K - kc), (unsigned)(Nn - n0), (unsigned long)lbr);
    };

    // Manual stage for the NN ("K x N") B case: f16 v4 loads along n, transpose.
    long offBn[4]; int bN[4], bK[4]; v4h_t brh[4];
#pragma unroll
    for (int i = 0; i < 4; ++i) {
        int lin = t + i * 256;
        int k = lin >> 4, n = (lin & 15) * 4;
        bN[i] = n; bK[i] = k;
        offBn[i] = (long)k * lbc + n0 + n;
    }
    auto brload = [&](int kc) {
#pragma unroll
        for (int i = 0; i < 4; ++i)
            brh[i] = *reinterpret_cast<const v4h_t*>(Bb + offBn[i] + (long)kc * lbc);
    };
    auto stageBnn = [&](int buf) {
#pragma unroll
        for (int i = 0; i < 4; ++i) {
            int n = bN[i], k = bK[i];
            Bs[buf][n + 0][k] = brh[i][0];
            Bs[buf][n + 1][k] = brh[i][1];
            Bs[buf][n + 2][k] = brh[i][2];
            Bs[buf][n + 3][k] = brh[i][3];
        }
    };

    const v8f_t vzero = {0.f, 0.f, 0.f, 0.f, 0.f, 0.f, 0.f, 0.f};
    v8f_t acc[4];
#pragma unroll
    for (int i = 0; i < 4; ++i) acc[i] = vzero;

    union afu { v16h_t v; v8h_t h[2]; };
    auto compute = [&](int buf) {
#pragma unroll
        for (int ks = 0; ks < 2; ++ks) {
            v16h_t bf = *reinterpret_cast<const v16h_t*>(
                &Bs[buf][wn * 16 + c16][ks * 32 + kh * 16]);
            afu af[4];
#pragma unroll
            for (int mt = 0; mt < 4; ++mt) {
                const _Float16* ap = &As[buf][wm * 64 + mt * 16 + c16][ks * 32];
                af[mt].h[0] = *reinterpret_cast<const v8h_t*>(ap + kh * 16);
                af[mt].h[1] = *reinterpret_cast<const v8h_t*>(ap + kh * 16 + 8);
            }
#pragma unroll
            for (int mt = 0; mt < 4; ++mt)
                acc[mt] = __builtin_amdgcn_wmma_f32_16x16x32_f16(
                    false, af[mt].v, false, bf, (short)0, acc[mt], false, false);
        }
    };

    // ---- pipelined main loop --------------------------------------------
    if (wave == 0) issueA(0, 0);
    if (bnt) { if (wave == 1) issueB(0, 0); }
    else { brload(0); stageBnn(0); }
    if (wave <= 1) __builtin_amdgcn_s_wait_tensorcnt(0);
    __syncthreads();

    int buf = 0;
    for (int kc = 64; kc < K; kc += 64) {
        if (wave == 0) issueA(buf ^ 1, kc);          // async DMA into other buffer
        if (bnt) { if (wave == 1) issueB(buf ^ 1, kc); }
        else brload(kc);
        compute(buf);                                 // 8 WMMAs overlap the DMA
        if (!bnt) stageBnn(buf ^ 1);
        if (wave <= 1) __builtin_amdgcn_s_wait_tensorcnt(0);
        __syncthreads();
        buf ^= 1;
    }
    compute(buf);

    // C layout: VGPR r -> M = r + 8*kh, N = c16
    const long n = n0 + wn * 16 + c16;
    const float bv = bias ? bias[n] : 0.0f;
    if (c_half) {
        _Float16* Ch = (_Float16*)Cp;
#pragma unroll
        for (int mt = 0; mt < 4; ++mt)
#pragma unroll
            for (int r = 0; r < 8; ++r) {
                long m = m0 + wm * 64 + mt * 16 + kh * 8 + r;
                if (m < M) Ch[zC + m * ldc + n] = (_Float16)(alpha * acc[mt][r] + bv);
            }
    } else {
        float* Cf = (float*)Cp;
#pragma unroll
        for (int mt = 0; mt < 4; ++mt)
#pragma unroll
            for (int r = 0; r < 8; ++r) {
                long m = m0 + wm * 64 + mt * 16 + kh * 8 + r;
                if (m < M) Cf[zC + m * ldc + n] = alpha * acc[mt][r] + bv;
            }
    }
}

// ---------------------------------------------------------------------------
__global__ void k_f32_to_f16(const float* __restrict__ src, _Float16* __restrict__ dst,
                             long n4)
{
    long i = (long)blockIdx.x * 256 + threadIdx.x;
    if (i < n4) {
        float4 v = reinterpret_cast<const float4*>(src)[i];
        v4h_t h4 = {(_Float16)v.x, (_Float16)v.y, (_Float16)v.z, (_Float16)v.w};
        reinterpret_cast<v4h_t*>(dst)[i] = h4;
    }
}

// Embedding gather -> f16 query input and f16 gru_input[..., :H].
__global__ void k_embed(const int* __restrict__ prev, const float* __restrict__ emb,
                        const float* __restrict__ h0,
                        _Float16* __restrict__ qin_h, _Float16* __restrict__ gru_in_h)
{
    long i4 = (long)blockIdx.x * 256 + threadIdx.x;  // < B*S*H/4
    long i  = i4 * 4;
    int  j  = (int)(i & (HH - 1));
    long bs = i >> 9;
    int  b  = (int)(bs >> 7);
    float4 e = *reinterpret_cast<const float4*>(&emb[(long)prev[bs] * HH + j]);
    v4h_t eh = {(_Float16)e.x, (_Float16)e.y, (_Float16)e.z, (_Float16)e.w};
    *reinterpret_cast<v4h_t*>(&gru_in_h[bs * (2 * HH) + j]) = eh;
    float4 h = *reinterpret_cast<const float4*>(&h0[(long)b * HH + j]);
    v4h_t qh = {(_Float16)(e.x + h.x), (_Float16)(e.y + h.y),
                (_Float16)(e.z + h.z), (_Float16)(e.w + h.w)};
    *reinterpret_cast<v4h_t*>(&qin_h[i]) = qh;
}

// Row softmax over 256 keys: f32 scores in -> f16 probabilities out.
__global__ void k_softmax256(const float* __restrict__ x, _Float16* __restrict__ ph)
{
    __shared__ float red[256];
    long row = blockIdx.x;
    int t = threadIdx.x;
    float v = x[row * SRCL + t];
    red[t] = v; __syncthreads();
    for (int o = 128; o > 0; o >>= 1) { if (t < o) red[t] = fmaxf(red[t], red[t + o]); __syncthreads(); }
    float m = red[0]; __syncthreads();
    float e = __expf(v - m);
    red[t] = e; __syncthreads();
    for (int o = 128; o > 0; o >>= 1) { if (t < o) red[t] += red[t + o]; __syncthreads(); }
    ph[row * SRCL + t] = (_Float16)(e / red[0]);
}

// One GRU step: elementwise gates; writes f32 state + f16 copies for GEMMs.
__global__ void k_gru_step(int s, const float* __restrict__ gi, const float* __restrict__ gh,
                           float* __restrict__ h, _Float16* __restrict__ h_h,
                           float* __restrict__ hid_out, _Float16* __restrict__ hid_h)
{
    int i = blockIdx.x * 256 + threadIdx.x;          // < B*H
    int b = i >> 9, j = i & (HH - 1);
    const float* gip = gi + ((long)(b * SS + s)) * (3 * HH);
    const float* ghp = gh + (long)b * (3 * HH);
    float r = 1.0f / (1.0f + __expf(-(gip[j] + ghp[j])));
    float z = 1.0f / (1.0f + __expf(-(gip[HH + j] + ghp[HH + j])));
    float n = tanhf(gip[2 * HH + j] + r * ghp[2 * HH + j]);
    float hn = (1.0f - z) * n + z * h[i];
    h[i] = hn;
    h_h[i] = (_Float16)hn;
    long o = ((long)(b * SS + s)) * HH + j;
    hid_out[o] = hn;
    hid_h[o] = (_Float16)hn;
}

// Vocab softmax (rows of length V=8000), in place on d_out.
__global__ void k_softmax_vocab(float* __restrict__ x)
{
    __shared__ float red[256];
    long row = blockIdx.x;
    int t = threadIdx.x;
    float* p = x + row * VV;
    float vals[32];
    float m = -3.4e38f;
#pragma unroll
    for (int i = 0; i < 32; ++i) {
        int idx = t + i * 256;
        float v = (idx < VV) ? p[idx] : -3.4e38f;
        vals[i] = v; m = fmaxf(m, v);
    }
    red[t] = m; __syncthreads();
    for (int o = 128; o > 0; o >>= 1) { if (t < o) red[t] = fmaxf(red[t], red[t + o]); __syncthreads(); }
    m = red[0]; __syncthreads();
    float s = 0.0f;
#pragma unroll
    for (int i = 0; i < 32; ++i) {
        int idx = t + i * 256;
        if (idx < VV) { vals[i] = __expf(vals[i] - m); s += vals[i]; }
    }
    red[t] = s; __syncthreads();
    for (int o = 128; o > 0; o >>= 1) { if (t < o) red[t] += red[t + o]; __syncthreads(); }
    float inv = 1.0f / red[0];
#pragma unroll
    for (int i = 0; i < 32; ++i) {
        int idx = t + i * 256;
        if (idx < VV) p[idx] = vals[i] * inv;
    }
}

// ---------------------------------------------------------------------------
static inline void gemm(hipStream_t st, const _Float16* A, const _Float16* B,
                        const float* bias, void* C, int c_half, int M, int N, int K,
                        long lar, long lbr, long lbc, long ldc,
                        int batch, int binner, long sAo, long sAi, long sBo, long sBi,
                        long sCo, long sCi, float alpha)
{
    dim3 g((unsigned)(N / 64), (unsigned)((M + 127) / 128), (unsigned)batch);
    k_gemm_wmma<<<g, dim3(256), 0, st>>>(A, B, bias, C, c_half, M, N, K,
                                         lar, lbr, lbc, ldc,
                                         binner, sAo, sAi, sBo, sBi, sCo, sCi, alpha);
}

static inline void cvt16(hipStream_t st, const float* src, _Float16* dst, long n)
{
    long n4 = n / 4;
    k_f32_to_f16<<<(unsigned)((n4 + 255) / 256), 256, 0, st>>>(src, dst, n4);
}

extern "C" void kernel_launch(void* const* d_in, const int* in_sizes, int n_in,
                              void* d_out, int out_size, void* d_ws, size_t ws_size,
                              hipStream_t stream)
{
    const float* enc        = (const float*)d_in[0];
    const int*   prev       = (const int*)  d_in[1];
    const float* h0         = (const float*)d_in[2];
    const float* embedding  = (const float*)d_in[3];
    const float* in_proj_w  = (const float*)d_in[4];
    const float* in_proj_b  = (const float*)d_in[5];
    const float* out_proj_w = (const float*)d_in[6];
    const float* out_proj_b = (const float*)d_in[7];
    const float* gru_w_ih   = (const float*)d_in[8];
    const float* gru_w_hh   = (const float*)d_in[9];
    const float* gru_b_ih   = (const float*)d_in[10];
    const float* gru_b_hh   = (const float*)d_in[11];
    const float* proj_w     = (const float*)d_in[12];
    const float* proj_b     = (const float*)d_in[13];

    // ---- workspace layout (byte cursor) ----------------------------------
    char* cur = (char*)d_ws;
    auto alloc = [&](size_t bytes) -> void* {
        void* r = (void*)cur; cur += (bytes + 255) & ~(size_t)255; return r;
    };
    const long N_BSH = (long)BB * SS * HH;
    const long N_BLH = (long)BB * SRCL * HH;
    const long N_SC  = (long)BB * NHH * SS * SRCL;

    float*    scores   = (float*)   alloc(N_SC * 4);          // also reused as gi
    float*    gi       = scores;                               // [B,S,3H] f32 alias
    float*    ghbuf    = (float*)   alloc((long)BB * 3 * HH * 4);
    float*    hbuf     = (float*)   alloc((long)BB * HH * 4);
    _Float16* enc_h    = (_Float16*)alloc(N_BLH * 2);
    _Float16* qin_h    = (_Float16*)alloc(N_BSH * 2);
    _Float16* gru_in_h = (_Float16*)alloc(2 * N_BSH * 2);
    _Float16* qb_h     = (_Float16*)alloc(N_BSH * 2);
    _Float16* kb_h     = (_Float16*)alloc(N_BLH * 2);
    _Float16* vb_h     = (_Float16*)alloc(N_BLH * 2);
    _Float16* P_h      = (_Float16*)alloc(N_SC * 2);
    _Float16* ctx_h    = (_Float16*)alloc(N_BSH * 2);
    _Float16* h_h      = (_Float16*)alloc((long)BB * HH * 2);
    _Float16* hid_h    = (_Float16*)alloc(N_BSH * 2);
    _Float16* ipw_h    = (_Float16*)alloc((long)3 * HH * HH * 2);
    _Float16* opw_h    = (_Float16*)alloc((long)HH * HH * 2);
    _Float16* wih_h    = (_Float16*)alloc((long)3 * HH * 2 * HH * 2);
    _Float16* whh_h    = (_Float16*)alloc((long)3 * HH * HH * 2);
    _Float16* pw_h     = (_Float16*)alloc((long)VV * HH * 2);

    float* probs      = (float*)d_out;                     // [B,S,V]
    float* out_hidden = probs + (long)BB * SS * VV;        // [B,S,H]

    // 0) one-shot f32 -> f16 conversions (weights + encoder features)
    cvt16(stream, enc,        enc_h, N_BLH);
    cvt16(stream, in_proj_w,  ipw_h, (long)3 * HH * HH);
    cvt16(stream, out_proj_w, opw_h, (long)HH * HH);
    cvt16(stream, gru_w_ih,   wih_h, (long)3 * HH * 2 * HH);
    cvt16(stream, gru_w_hh,   whh_h, (long)3 * HH * HH);
    cvt16(stream, proj_w,     pw_h,  (long)VV * HH);

    // 1) embedding gather + query input (+ first half of gru_input), f16
    k_embed<<<(unsigned)(N_BSH / 4 / 256), 256, 0, stream>>>(prev, embedding, h0,
                                                             qin_h, gru_in_h);

    // 2) q/k/v projections
    gemm(stream, qin_h, ipw_h,                  in_proj_b,          qb_h, 1,
         BB * SS,   HH, HH, HH, HH, 1, HH, 1, 1, 0, 0, 0, 0, 0, 0, 1.0f);
    gemm(stream, enc_h, ipw_h + (long)HH * HH,  in_proj_b + HH,     kb_h, 1,
         BB * SRCL, HH, HH, HH, HH, 1, HH, 1, 1, 0, 0, 0, 0, 0, 0, 1.0f);
    gemm(stream, enc_h, ipw_h + 2L * HH * HH,   in_proj_b + 2 * HH, vb_h, 1,
         BB * SRCL, HH, HH, HH, HH, 1, HH, 1, 1, 0, 0, 0, 0, 0, 0, 1.0f);

    // 3) scores[b,h] = (q_bh @ k_bh^T) / 8; batch z = b*NH + h; f32 out
    gemm(stream, qb_h, kb_h, nullptr, scores, 0,
         SS, SRCL, HDD, HH, HH, 1, SRCL,
         BB * NHH, NHH,
         (long)SS * HH, HDD,
         (long)SRCL * HH, HDD,
         (long)NHH * SS * SRCL, (long)SS * SRCL,
         0.125f);

    // 4) softmax over keys -> f16 P
    k_softmax256<<<(unsigned)(BB * NHH * SS), 256, 0, stream>>>(scores, P_h);

    // 5) ctx = P @ V (NN path, lbc = H); f16 out
    gemm(stream, P_h, vb_h, nullptr, ctx_h, 1,
         SS, HDD, SRCL, SRCL, 0, /*lbc=*/HH, /*ldc=*/HH,
         BB * NHH, NHH,
         (long)NHH * SS * SRCL, (long)SS * SRCL,
         (long)SRCL * HH, HDD,
         (long)SS * HH, HDD,
         1.0f);

    // 6) out_proj -> gru_input[..., H:2H] (f16, ldc = 2H)
    gemm(stream, ctx_h, opw_h, out_proj_b, gru_in_h + HH, 1,
         BB * SS, HH, HH, HH, HH, 1, 2 * HH, 1, 1, 0, 0, 0, 0, 0, 0, 1.0f);

    // 7) gi = gru_input @ W_ih^T + b_ih (f32, reuses scores region)
    gemm(stream, gru_in_h, wih_h, gru_b_ih, gi, 0,
         BB * SS, 3 * HH, 2 * HH, 2 * HH, 2 * HH, 1, 3 * HH,
         1, 1, 0, 0, 0, 0, 0, 0, 1.0f);

    // 8) recurrent scan
    (void)hipMemcpyAsync(hbuf, h0, (size_t)BB * HH * sizeof(float),
                         hipMemcpyDeviceToDevice, stream);
    cvt16(stream, h0, h_h, (long)BB * HH);
    for (int s = 0; s < SS; ++s) {
        gemm(stream, h_h, whh_h, gru_b_hh, ghbuf, 0,
             BB, 3 * HH, HH, HH, HH, 1, 3 * HH,
             1, 1, 0, 0, 0, 0, 0, 0, 1.0f);
        k_gru_step<<<(unsigned)(BB * HH / 256), 256, 0, stream>>>(
            s, gi, ghbuf, hbuf, h_h, out_hidden, hid_h);
    }

    // 9) logits = hidden @ proj_w^T + proj_b -> d_out (f32)
    gemm(stream, hid_h, pw_h, proj_b, probs, 0,
         BB * SS, VV, HH, HH, HH, 1, VV,
         1, 1, 0, 0, 0, 0, 0, 0, 1.0f);

    // 10) vocab softmax in place
    k_softmax_vocab<<<(unsigned)(BB * SS), 256, 0, stream>>>(probs);
}
